// NewNodeFeaturesFrom3Body_52707838657165
// MI455X (gfx1250) — matrix operand order repaired and checked by reading, not compiled, not run on hardware
//
#include <hip/hip_runtime.h>
#include <hip/hip_bf16.h>

typedef __attribute__((ext_vector_type(16))) __bf16 v16bf;
typedef __attribute__((ext_vector_type(8)))  float  v8f;

#define NODES_PER_BLK 16
#define CCH   64          // channels
#define DIRR  9           // irreps dim (1+3+5)
#define C3    192         // 3*CCH
#define KDIM  30          // 3*LZ
#define KPAD  32
#define WCOLS 1728        // DIRR*C3 (W_chem row stride)
#define THREADS 384       // 12 waves * 32 lanes
#define WSB_ELEMS (WCOLS * KPAD)   // transposed bf16 W_chem: [col][k]

__device__ __forceinline__ unsigned short f2bfbits(float f) {
    return __builtin_bit_cast(unsigned short, (__bf16)f);   // hardware cvt
}
__device__ __forceinline__ __bf16 bfbits(unsigned short u) {
    return __builtin_bit_cast(__bf16, u);
}

// ---- prep: W_chem (30x1728 f32, row-major) -> ws (1728x32 bf16, col-major, K-padded) ----
__global__ __launch_bounds__(256)
void wchem_prep_kernel(const float* __restrict__ W_chem, unsigned short* __restrict__ wsT)
{
    int idx = blockIdx.x * 256 + threadIdx.x;     // col*32 + k
    if (idx >= WSB_ELEMS) return;
    int col = idx >> 5;
    int k   = idx & 31;
    float v = (k < KDIM) ? W_chem[(size_t)k * WCOLS + col] : 0.f;
    wsT[idx] = f2bfbits(v);
}

template <bool PRE>
__global__ __launch_bounds__(THREADS)
void fused_3body_kernel(const float* __restrict__ node_nbody,
                        const float* __restrict__ node_attr,
                        const float* __restrict__ W_chem,
                        const unsigned short* __restrict__ wsT,
                        const float* __restrict__ W_body,
                        float* __restrict__ out,
                        int N)
{
    __shared__ unsigned short lA[NODES_PER_BLK][KPAD];      // nab tile, bf16 bits
    __shared__ float obuf[NODES_PER_BLK * CCH * DIRR];      // 36 KB accumulator

    const int tid  = threadIdx.x;
    const int lane = tid & 31;
    const int wave = tid >> 5;          // 0..11
    const int hi   = lane >> 4;         // lane group 0/1
    const int lo   = lane & 15;
    const int nodeBase = blockIdx.x * NODES_PER_BLK;
    const int LN = 10 * N;              // L * N

    // ---- zero the per-block output accumulator ----
    for (int i = tid; i < NODES_PER_BLK * CCH * DIRR; i += THREADS) obuf[i] = 0.f;

    // ---- build scrambled nab tile (faithful torch reshape), scaled 1/sqrt(30), as bf16 ----
    const float invs30 = 0.18257418583505536f;
    for (int idx = tid; idx < NODES_PER_BLK * KPAD; idx += THREADS) {
        int row = idx >> 5;             // node within tile
        int k   = idx & 31;
        float v = 0.f;
        if (k < KDIM) {
            long long flat = (long long)(nodeBase + row) * KDIM + k;
            int b   = (int)(flat / LN);
            int rem = (int)(flat - (long long)b * LN);
            int l   = rem / N;
            int nn  = rem - l * N;
            v = node_attr[nn * 10 + l] * invs30;
        }
        lA[row][k] = f2bfbits(v);
    }
    __syncthreads();

    // ---- A fragment: 16x32 bf16, contiguous k order (lane M=lo; element s: K = 16*hi + s) ----
    v16bf afrag;
    #pragma unroll
    for (int v = 0; v < 8; ++v) {
        int kb = hi * 16 + 2 * v;
        afrag[2 * v]     = bfbits(lA[lo][kb]);
        afrag[2 * v + 1] = bfbits(lA[lo][kb + 1]);
    }

    const int c3base = wave * 16;

    // ---- 9 WMMAs: wz[node][c3][d] for this wave's 16 c3 columns ----
    v8f dfr[DIRR];
    #pragma unroll
    for (int d = 0; d < DIRR; ++d) {
        v16bf bfrag;                                         // 32x16 bf16 B layout
        if constexpr (PRE) {
            // fragment = 16 consecutive K values at fixed column -> one 32B read
            const unsigned short* p =
                wsT + ((size_t)(d * C3 + c3base + lo) << 5) + hi * 16;
            bfrag = *(const v16bf*)p;                        // 2x global_load_b128
        } else {
            const float* wcol = W_chem + d * C3 + c3base + lo;
            #pragma unroll
            for (int v = 0; v < 8; ++v) {
                int k0 = hi * 16 + 2 * v;
                int ka = (k0     < KDIM) ? k0     : 0;       // clamp: branch-free loads
                int kb2= (k0 + 1 < KDIM) ? k0 + 1 : 0;
                float f0 = wcol[(size_t)ka  * WCOLS];
                float f1 = wcol[(size_t)kb2 * WCOLS];
                f0 = (k0     < KDIM) ? f0 : 0.f;
                f1 = (k0 + 1 < KDIM) ? f1 : 0.f;
                bfrag[2 * v]     = (__bf16)f0;
                bfrag[2 * v + 1] = (__bf16)f1;
            }
        }
        v8f acc = {0.f, 0.f, 0.f, 0.f, 0.f, 0.f, 0.f, 0.f};
        dfr[d] = __builtin_amdgcn_wmma_f32_16x16x32_bf16(
            false, afrag, false, bfrag, (short)0, acc, false, false);
    }

    // ---- Wigner-3j contraction: lane owns c3 = c3base+lo; reg r owns node hi*8+r ----
    const int   c3 = c3base + lo;
    const int   b  = c3 % 3;
    const int   cc = c3 / 3;
    const float wb = W_body[b] * 0.5773502691896258f;        // / sqrt(3)

    const float K1 = 0.5773502691896258f;   // 1/sqrt(3)
    const float K2 = 0.4472135954999579f;   // 1/sqrt(5)
    const float S  = 0.31622776601683794f;  // 1/sqrt(10)
    const float P  = 0.3651483716701107f;   // 2/sqrt(30)
    const float Q  = 0.18257418583505536f;  // 1/sqrt(30)
    const float T1 = 0.2390457218668787f;   // sqrt(2/35)
    const float T2 = 0.11952286093343935f;  // sqrt(2/35)/2
    const float T3 = 0.20701966780270626f;  // sqrt(3/70)

    #pragma unroll
    for (int r = 0; r < 8; ++r) {
        const int m = hi * 8 + r;
        const float* xp = node_nbody + ((((size_t)b * N + (nodeBase + m)) * CCH + cc) * DIRR);
        float x0 = __builtin_nontemporal_load(xp + 0);
        float x1 = __builtin_nontemporal_load(xp + 1);
        float x2 = __builtin_nontemporal_load(xp + 2);
        float x3 = __builtin_nontemporal_load(xp + 3);
        float x4 = __builtin_nontemporal_load(xp + 4);
        float x5 = __builtin_nontemporal_load(xp + 5);
        float x6 = __builtin_nontemporal_load(xp + 6);
        float x7 = __builtin_nontemporal_load(xp + 7);
        float x8 = __builtin_nontemporal_load(xp + 8);

        float w0 = dfr[0][r], w1 = dfr[1][r], w2 = dfr[2][r], w3 = dfr[3][r],
              w4 = dfr[4][r], w5 = dfr[5][r], w6 = dfr[6][r], w7 = dfr[7][r],
              w8 = dfr[8][r];

        // l3 = 0
        float m0 = w0*x0 + K1*(w1*x1 + w2*x2 + w3*x3)
                 + K2*(w4*x4 + w5*x5 + w6*x6 + w7*x7 + w8*x8);
        // l3 = 1  (0x1->1, 1x0->1, 1x2->1, 2x1->1)
        float m1 = K1*(w0*x1 + w1*x0)
                 + S*(w3*x4 + w2*x5) - Q*(w1*x6) - S*(w1*x8)
                 + S*(w4*x3 + w5*x2) - Q*(w6*x1) - S*(w8*x1);
        float m2 = K1*(w0*x2 + w2*x0)
                 + S*(w1*x5) + P*(w2*x6) + S*(w3*x7)
                 + S*(w5*x1) + P*(w6*x2) + S*(w7*x3);
        float m3 = K1*(w0*x3 + w3*x0)
                 + S*(w1*x4) - Q*(w3*x6) + S*(w2*x7) + S*(w3*x8)
                 + S*(w4*x1) - Q*(w6*x3) + S*(w7*x2) + S*(w8*x3);
        // l3 = 2  (0x2->2, 2x0->2, 1x1->2, 2x2->2)
        float m4 = K2*(w0*x4 + w4*x0)
                 + S*(w3*x1 + w1*x3)
                 - T1*(w6*x4 + w4*x6) + T3*(w5*x7 + w7*x5);
        float m5 = K2*(w0*x5 + w5*x0)
                 + S*(w1*x2 + w2*x1)
                 + T2*(w6*x5 + w5*x6) - T3*(w8*x5 + w5*x8) + T3*(w4*x7 + w7*x4);
        float m6 = K2*(w0*x6 + w6*x0)
                 + P*(w2*x2) - Q*(w1*x1) - Q*(w3*x3)
                 + T1*(w6*x6 - w4*x4 - w8*x8) + T2*(w5*x5 + w7*x7);
        float m7 = K2*(w0*x7 + w7*x0)
                 + S*(w3*x2 + w2*x3)
                 + T2*(w6*x7 + w7*x6) + T3*(w8*x7 + w7*x8) + T3*(w4*x5 + w5*x4);
        float m8 = K2*(w0*x8 + w8*x0)
                 + S*(w3*x3 - w1*x1)
                 - T1*(w6*x8 + w8*x6) - T3*(w5*x5) + T3*(w7*x7);

        // body-order reduction: 3 lanes (b=0,1,2) fold into out[node][c][*]
        float* ob = &obuf[(m * CCH + cc) * DIRR];
        atomicAdd(&ob[0], m0 * wb);
        atomicAdd(&ob[1], m1 * wb);
        atomicAdd(&ob[2], m2 * wb);
        atomicAdd(&ob[3], m3 * wb);
        atomicAdd(&ob[4], m4 * wb);
        atomicAdd(&ob[5], m5 * wb);
        atomicAdd(&ob[6], m6 * wb);
        atomicAdd(&ob[7], m7 * wb);
        atomicAdd(&ob[8], m8 * wb);
    }
    __syncthreads();

    // ---- coalesced non-temporal store of this block's 16 nodes ----
    float* obase = out + (size_t)nodeBase * CCH * DIRR;
    for (int i = tid; i < NODES_PER_BLK * CCH * DIRR; i += THREADS)
        __builtin_nontemporal_store(obuf[i], obase + i);
}

extern "C" void kernel_launch(void* const* d_in, const int* in_sizes, int n_in,
                              void* d_out, int out_size, void* d_ws, size_t ws_size,
                              hipStream_t stream) {
    const float* node_nbody = (const float*)d_in[0];
    const float* node_attr  = (const float*)d_in[1];
    const float* W_chem     = (const float*)d_in[2];
    const float* W_body     = (const float*)d_in[3];
    float* out = (float*)d_out;

    const int N = in_sizes[1] / 10;            // 20000
    const int blocks = N / NODES_PER_BLK;      // 1250 (N divisible by 16)

    const size_t ws_needed = (size_t)WSB_ELEMS * sizeof(unsigned short); // ~108 KB
    if (ws_size >= ws_needed) {
        unsigned short* wsT = (unsigned short*)d_ws;
        wchem_prep_kernel<<<dim3((WSB_ELEMS + 255) / 256), dim3(256), 0, stream>>>(W_chem, wsT);
        fused_3body_kernel<true><<<dim3(blocks), dim3(THREADS), 0, stream>>>(
            node_nbody, node_attr, W_chem, wsT, W_body, out, N);
    } else {
        fused_3body_kernel<false><<<dim3(blocks), dim3(THREADS), 0, stream>>>(
            node_nbody, node_attr, W_chem, (const unsigned short*)nullptr, W_body, out, N);
    }
}
